// HybridSelfAttentionBlockWithDisplacement_26774826123571
// MI455X (gfx1250) — matrix operand order repaired and checked by reading, not compiled, not run on hardware
//
#include <hip/hip_runtime.h>
#include <hip/hip_bf16.h>

// ---------------------------------------------------------------------------
// HybridSelfAttentionBlockWithDisplacement for MI455X (gfx1250, wave32, WMMA)
//
// Shapes: B=2 LS=1024 G=32 D=512 H=8 DH=64 K=32 PE=64 INNER=512 HID=128
//
//  * LayerNorm -> f16 activations
//  * Dense projections via v_wmma_f32_16x16x32_f16, f32 accumulate.
//    Weights are repacked once into per-lane WMMA fragment layout so each
//    B fragment is a single aligned 32B load (2x global_load_b128).
//    Each wave computes a 16x64 strip: 4 WMMAs per K-step sharing one A frag.
//  * ctx@Wk / ctx@Wv factored: feature part precomputed per source row
//    (L2-resident Ks/Vs), RPE part folded onto Q side (Tq) for logits and
//    onto attention-weight side (r-bar @ Wv_rpe) for the output.
//  * Fused local-attention kernel; online-softmax global attention;
//    WMMA MLP GEMM + fused gelu/LN/W2/tanh epilogue.
// ---------------------------------------------------------------------------

constexpr int cB = 2, cLS = 1024, cG = 32, cD = 512, cH = 8, cDH = 64;
constexpr int cK = 32, cPE = 64, cINNER = 512, cHID = 128;
constexpr float cSCALE = 0.125f;   // DH^-0.5
constexpr float cMAXD = 5.0f;

typedef __attribute__((ext_vector_type(16))) _Float16 v16h;
typedef __attribute__((ext_vector_type(8)))  float    v8f;

// ---------------------------------------------------------------- weight pack
// Fused f32->f16 conversion + repack of W[Kd,N] into WMMA B-fragment order:
// Bp[((kb*nblk + nb)*32 + lane)*16 + e] = W[kb*32 + ((lane&16)?16:0) + e,
//                                          nb*16 + (lane&15)]
__global__ void pack_w_kernel(const float* __restrict__ W,
                              _Float16* __restrict__ Bp, int Kd, int N) {
  int o = blockIdx.x * blockDim.x + threadIdx.x;
  if (o >= Kd * N) return;
  int e   = o & 15;
  int L   = (o >> 4) & 31;
  int blk = o >> 9;
  int nblk = N >> 4;
  int nb = blk % nblk;
  int kb = blk / nblk;
  int n = (nb << 4) + (L & 15);
  int k = (kb << 5) + ((L & 16) ? 16 : 0) + e;
  Bp[o] = (_Float16)W[(size_t)k * N + n];
}

// ---------------------------------------------------------------- layernorm
__global__ void ln_f16_kernel(const float* __restrict__ x,
                              const float* __restrict__ g,
                              const float* __restrict__ b,
                              _Float16* __restrict__ y, int cols) {
  int row = blockIdx.x, tid = threadIdx.x;
  const float* xr = x + (size_t)row * cols;
  float s = 0.f, ss = 0.f;
  for (int j = tid; j < cols; j += blockDim.x) { float v = xr[j]; s += v; ss += v * v; }
  __shared__ float r1[256], r2[256];
  r1[tid] = s; r2[tid] = ss; __syncthreads();
  for (int o = blockDim.x >> 1; o > 0; o >>= 1) {
    if (tid < o) { r1[tid] += r1[tid + o]; r2[tid] += r2[tid + o]; }
    __syncthreads();
  }
  float mean = r1[0] / cols;
  float var  = r2[0] / cols - mean * mean;
  float inv  = rsqrtf(var + 1e-5f);
  for (int j = tid; j < cols; j += blockDim.x) {
    float v = (xr[j] - mean) * inv * g[j] + b[j];
    y[(size_t)row * cols + j] = (_Float16)v;
  }
}

// ---------------------------------------------------------------- WMMA GEMM
// C[M,N] = A[M,Kd] @ W[Kd,N] with W pre-packed (pack_w_kernel).
// Each wave: one 16-row x 64-col strip = 4 accumulators; 4 WMMA per K-step.
__global__ void gemm_wmma_kernel(const _Float16* __restrict__ A, int lda,
                                 const _Float16* __restrict__ Bp, int nblk,
                                 float* __restrict__ C, int ldc,
                                 int M, int N, int Kd,
                                 const float* __restrict__ bias,
                                 const float* __restrict__ resid, int ldr,
                                 _Float16* __restrict__ Ch, int ldh) {
  const int lane = threadIdx.x & 31;
  const int wave = threadIdx.x >> 5;
  const int stripsN = N >> 6;                 // 64-wide strips
  const int strip = blockIdx.x * (blockDim.x >> 5) + wave;
  if (strip >= (M >> 4) * stripsN) return;
  const int tm  = (strip / stripsN) << 4;
  const int nb0 = (strip % stripsN) << 2;     // first 16-col block of strip

  const int mrow  = tm + (lane & 15);
  const int koffA = (lane & 16) ? 16 : 0;

  v8f acc[4];
#pragma unroll
  for (int j = 0; j < 4; ++j) acc[j] = v8f{};

  for (int k0 = 0; k0 < Kd; k0 += 32) {
    const int kb = k0 >> 5;
    // A fragment: 16 contiguous k (k = 16*(lane/16) + e), matching packed B
    v16h a;
    const _Float16* Ap = A + (size_t)mrow * lda + (k0 + koffA);
#pragma unroll
    for (int i = 0; i < 8; ++i) {
      int kk = 2 * i;
      a[2 * i]     = Ap[kk];
      a[2 * i + 1] = Ap[kk + 1];
    }
    if (k0 + 32 < Kd) {                        // prefetch next K-step
      __builtin_prefetch((const void*)(Ap + 32), 0, 1);
      __builtin_prefetch(
          (const void*)(Bp + ((size_t)((kb + 1) * nblk + nb0) * 32 + lane) * 16),
          0, 1);
    }
#pragma unroll
    for (int j = 0; j < 4; ++j) {
      const int nb = nb0 + j;
      v16h bf = *(const v16h*)(Bp + ((size_t)(kb * nblk + nb) * 32 + lane) * 16);
      acc[j] = __builtin_amdgcn_wmma_f32_16x16x32_f16(false, a, false, bf,
                                                      (short)0, acc[j],
                                                      false, false);
    }
  }
  const int nloc  = lane & 15;
  const int mbase = (lane & 16) ? 8 : 0;
#pragma unroll
  for (int j = 0; j < 4; ++j) {
#pragma unroll
    for (int r = 0; r < 8; ++r) {
      int m = tm + mbase + r;
      int n = ((nb0 + j) << 4) + nloc;
      float v = acc[j][r];
      if (bias)  v += bias[n];
      if (resid) v += resid[(size_t)m * ldr + n];
      C[(size_t)m * ldc + n] = v;
      if (Ch) Ch[(size_t)m * ldh + n] = (_Float16)v;
    }
  }
}

static inline void launch_gemm(hipStream_t s, const _Float16* A, int lda,
                               const _Float16* Bp, int N_for_nblk,
                               float* C, int ldc,
                               int M, int N, int Kd, const float* bias,
                               const float* resid, int ldr,
                               _Float16* Ch, int ldh) {
  int strips = (M / 16) * (N / 64);
  int blocks = (strips + 7) / 8;               // 8 waves (256 thr) per block
  gemm_wmma_kernel<<<blocks, 256, 0, s>>>(A, lda, Bp, N_for_nblk / 16, C, ldc,
                                          M, N, Kd, bias, resid, ldr, Ch, ldh);
}

// ---------------------------------------------------------------- Tq
// Tq[row, h*PE+p] = sum_d Wk_rpe[p, h*DH+d] * Q[row, h*DH+d]
__global__ void tq_kernel(const float* __restrict__ Q,
                          const float* __restrict__ Wk_rpe,
                          float* __restrict__ Tq) {
  int row = blockIdx.x, t = threadIdx.x;        // 512 threads
  int h = t >> 6, p = t & 63;
  const float* q = Q + (size_t)row * cINNER + h * cDH;
  const float* w = Wk_rpe + (size_t)p * cINNER + h * cDH;
  float acc = 0.f;
#pragma unroll
  for (int d = 0; d < cDH; ++d) acc += q[d] * w[d];
  Tq[(size_t)row * (cH * cPE) + t] = acc;
}

// ---------------------------------------------------------------- local attn
__global__ void local_attn_kernel(
    const float* __restrict__ Q, const float* __restrict__ Tq,
    const float* __restrict__ Ks, const float* __restrict__ Vs,
    const float* __restrict__ Kgl, const float* __restrict__ Vgl,
    const float* __restrict__ self_rpe, const float* __restrict__ rpe,
    const float* __restrict__ distances, const int* __restrict__ topk,
    const float* __restrict__ log_sigma, const float* __restrict__ gb_l,
    const float* __restrict__ head_weights, const float* __restrict__ positions,
    const float* __restrict__ Wv_rpe,
    _Float16* __restrict__ outh, float* __restrict__ attn_disp) {
  const int row = blockIdx.x;
  const int b = row / cLS;
  const int tid = threadIdx.x;                  // 256 threads

  __shared__ float sQ[cINNER];
  __shared__ float sT[cINNER];
  __shared__ const float* kp[65];
  __shared__ const float* vp[65];
  __shared__ float sl[cH][65];                  // logits -> attn (in place)
  __shared__ float srb[cH * cPE];               // r-bar
  __shared__ float swa[cK];
  __shared__ int   sidx[cK];

  if (tid < cK) sidx[tid] = topk[(size_t)row * cK + tid];
  for (int j = tid; j < cINNER; j += 256) {
    sQ[j] = Q[(size_t)row * cINNER + j];
    sT[j] = Tq[(size_t)row * cINNER + j];
  }
  __syncthreads();

  if (tid < 65) {
    int c = tid;
    const float *kq, *vq;
    if (c == 0)        { kq = Ks + (size_t)row * cINNER;
                         vq = Vs + (size_t)row * cINNER; }
    else if (c <= cK)  { size_t r = (size_t)(b * cLS + sidx[c - 1]) * cINNER;
                         kq = Ks + r; vq = Vs + r; }
    else               { size_t r = (size_t)(b * cG + (c - 1 - cK)) * cINNER;
                         kq = Kgl + r; vq = Vgl + r; }
    kp[c] = kq; vp[c] = vq;
  }
  __syncthreads();

  const float gbl = gb_l[0];
  for (int t = tid; t < cH * 65; t += 256) {
    int h = t / 65, c = t % 65;
    const float* kr = kp[c] + h * cDH;
    const float* qh = sQ + h * cDH;
    float dot = 0.f;
#pragma unroll
    for (int d = 0; d < cDH; ++d) dot += qh[d] * kr[d];
    float bias;
    if (c <= cK) {
      const float* rv = (c == 0) ? (self_rpe + (size_t)row * cPE)
                                 : (rpe + ((size_t)row * cK + (c - 1)) * cPE);
      const float* th = sT + h * cPE;
      float dr = 0.f;
#pragma unroll
      for (int p = 0; p < cPE; ++p) dr += th[p] * rv[p];
      dot += dr;
      float dloc = (c == 0) ? 0.f : distances[(size_t)row * cK + (c - 1)];
      float sig2 = __expf(2.f * log_sigma[h]);
      bias = -(dloc * dloc) / (2.f * sig2);
    } else {
      bias = gbl;
    }
    sl[h][c] = dot * cSCALE + bias;
  }
  __syncthreads();

  if (tid < cH) {                               // per-head softmax over 65
    float m = -1e30f;
    for (int c = 0; c < 65; ++c) m = fmaxf(m, sl[tid][c]);
    float s = 0.f;
    for (int c = 0; c < 65; ++c) { float e = __expf(sl[tid][c] - m); sl[tid][c] = e; s += e; }
    float inv = 1.f / s;
    for (int c = 0; c < 65; ++c) sl[tid][c] *= inv;
  }
  __syncthreads();

  // r-bar[h][p] = sum_{c<=K} attn[h][c] * rpe_cat[c][p]
  for (int t = tid; t < cH * cPE; t += 256) {
    int h = t >> 6, p = t & 63;
    float acc = sl[h][0] * self_rpe[(size_t)row * cPE + p];
    for (int c = 1; c <= cK; ++c)
      acc += sl[h][c] * rpe[((size_t)row * cK + (c - 1)) * cPE + p];
    srb[t] = acc;
  }
  __syncthreads();

  // out[j] = sum_c attn * V_feat  +  r-bar @ Wv_rpe
  for (int j = tid; j < cINNER; j += 256) {
    int h = j >> 6;
    float acc = 0.f;
    for (int c = 0; c < 65; ++c) acc += sl[h][c] * vp[c][j];
    const float* rb = srb + h * cPE;
    for (int p = 0; p < cPE; ++p) acc += rb[p] * Wv_rpe[(size_t)p * cINNER + j];
    outh[(size_t)row * cINNER + j] = (_Float16)acc;
  }

  // displacement from neighbor attention
  if (tid < cK) {
    float hw[cH], m = -1e30f;
#pragma unroll
    for (int h = 0; h < cH; ++h) { hw[h] = head_weights[h]; m = fmaxf(m, hw[h]); }
    float s = 0.f;
#pragma unroll
    for (int h = 0; h < cH; ++h) { hw[h] = __expf(hw[h] - m); s += hw[h]; }
    float w = 0.f;
#pragma unroll
    for (int h = 0; h < cH; ++h) w += (hw[h] / s) * sl[h][tid + 1];
    swa[tid] = w;
  }
  __syncthreads();
  if (tid == 0) {
    float s = 0.f;
    for (int k = 0; k < cK; ++k) s += swa[k];
    float inv = 1.f / (s + 1e-8f);
    float a0 = 0.f, a1 = 0.f;
    for (int k = 0; k < cK; ++k) {
      float w = swa[k] * inv;
      const float* pp = positions + (size_t)(b * cLS + sidx[k]) * 2;
      a0 += w * pp[0]; a1 += w * pp[1];
    }
    a0 -= positions[(size_t)row * 2 + 0];
    a1 -= positions[(size_t)row * 2 + 1];
    attn_disp[(size_t)row * 2 + 0] = a0;
    attn_disp[(size_t)row * 2 + 1] = a1;
  }
}

// ---------------------------------------------------------------- global attn
__global__ void global_attn_kernel(const float* __restrict__ Qg,
                                   const float* __restrict__ Ksg,
                                   const float* __restrict__ Kgg,
                                   const float* __restrict__ Vsg,
                                   const float* __restrict__ Vgg,
                                   const float* __restrict__ gb_g,
                                   _Float16* __restrict__ og) {
  int b = blockIdx.x / cH, h = blockIdx.x % cH;
  int g = threadIdx.x;                          // 32 threads = 1 wave
  const float* q = Qg + (size_t)(b * cG + g) * cINNER + h * cDH;
  float qr[cDH];
#pragma unroll
  for (int d = 0; d < cDH; ++d) qr[d] = q[d];
  float m = -1e30f, s = 0.f, acc[cDH];
#pragma unroll
  for (int d = 0; d < cDH; ++d) acc[d] = 0.f;
  const float gb = gb_g[0];
  for (int c = 0; c < cLS + cG; ++c) {
    const float* kr;
    const float* vr;
    if (c < cLS) { size_t r = (size_t)(b * cLS + c) * cINNER + h * cDH;
                   kr = Ksg + r; vr = Vsg + r; }
    else         { size_t r = (size_t)(b * cG + (c - cLS)) * cINNER + h * cDH;
                   kr = Kgg + r; vr = Vgg + r; }
    float dot = 0.f;
#pragma unroll
    for (int d = 0; d < cDH; ++d) dot += qr[d] * kr[d];
    float lg = dot * cSCALE + gb;
    float mn = fmaxf(m, lg);
    float corr = __expf(m - mn);
    float w = __expf(lg - mn);
    s = s * corr + w;
#pragma unroll
    for (int d = 0; d < cDH; ++d) acc[d] = acc[d] * corr + w * vr[d];
    m = mn;
  }
  float inv = 1.f / s;
  _Float16* o = og + (size_t)(b * cG + g) * cINNER + h * cDH;
  for (int d = 0; d < cDH; ++d) o[d] = (_Float16)(acc[d] * inv);
}

// ---------------------------------------------------------------- MLP tail
__global__ void mlp_disp_kernel(const float* __restrict__ h1,
                                const float* __restrict__ lng,
                                const float* __restrict__ lnb,
                                const float* __restrict__ W2,
                                const float* __restrict__ b2,
                                const float* __restrict__ attn_disp,
                                const float* __restrict__ log_attn_w,
                                const float* __restrict__ log_mlp_w,
                                float* __restrict__ disp) {
  int row = blockIdx.x, tid = threadIdx.x;      // 128 threads
  float x = h1[(size_t)row * cHID + tid];
  float ga = 0.5f * x * (1.0f + erff(x * 0.70710678118654752f));  // exact gelu
  __shared__ float r1[cHID], r2[cHID];
  r1[tid] = ga; r2[tid] = ga * ga; __syncthreads();
  for (int o = cHID >> 1; o > 0; o >>= 1) {
    if (tid < o) { r1[tid] += r1[tid + o]; r2[tid] += r2[tid + o]; }
    __syncthreads();
  }
  float mean = r1[0] / cHID, var = r2[0] / cHID - mean * mean;
  __syncthreads();
  float hn = (ga - mean) * rsqrtf(var + 1e-5f) * lng[tid] + lnb[tid];
  r1[tid] = hn * W2[tid * 2 + 0]; r2[tid] = hn * W2[tid * 2 + 1]; __syncthreads();
  for (int o = cHID >> 1; o > 0; o >>= 1) {
    if (tid < o) { r1[tid] += r1[tid + o]; r2[tid] += r2[tid + o]; }
    __syncthreads();
  }
  if (tid == 0) {
    float md0 = r1[0] + b2[0], md1 = r2[0] + b2[1];
    float wa = __expf(log_attn_w[0]), wm = __expf(log_mlp_w[0]);
    float inv = 1.f / (wa + wm + 1e-8f);
    float c0 = (wa * attn_disp[row * 2 + 0] + wm * md0) * inv;
    float c1 = (wa * attn_disp[row * 2 + 1] + wm * md1) * inv;
    disp[row * 2 + 0] = tanhf(c0 / cMAXD) * cMAXD;
    disp[row * 2 + 1] = tanhf(c1 / cMAXD) * cMAXD;
  }
}

// ---------------------------------------------------------------- driver
extern "C" void kernel_launch(void* const* d_in, const int* in_sizes, int n_in,
                              void* d_out, int out_size, void* d_ws, size_t ws_size,
                              hipStream_t stream) {
  (void)in_sizes; (void)n_in; (void)out_size; (void)ws_size;
  const float* spatial        = (const float*)d_in[0];
  const float* global_latents = (const float*)d_in[1];
  const float* positions      = (const float*)d_in[2];
  const float* rpe            = (const float*)d_in[3];
  const float* self_rpe       = (const float*)d_in[4];
  const float* distances      = (const float*)d_in[5];
  const float* ln_s_g = (const float*)d_in[6],  *ln_s_b = (const float*)d_in[7];
  const float* ln_g_g = (const float*)d_in[8],  *ln_g_b = (const float*)d_in[9];
  const float* Wq_l = (const float*)d_in[10];
  const float* Wk_l = (const float*)d_in[11];
  const float* Wv_l = (const float*)d_in[12];
  const float* Wo_l = (const float*)d_in[13];
  const float* bo_l = (const float*)d_in[14];
  const float* log_sigma = (const float*)d_in[15];
  const float* gb_l = (const float*)d_in[16];
  const float* Wq_g = (const float*)d_in[17];
  const float* Wk_g = (const float*)d_in[18];
  const float* Wv_g = (const float*)d_in[19];
  const float* Wo_g = (const float*)d_in[20];
  const float* bo_g = (const float*)d_in[21];
  const float* gb_g = (const float*)d_in[22];
  const float* head_weights = (const float*)d_in[23];
  const float* mlp_W1 = (const float*)d_in[24];
  const float* mlp_b1 = (const float*)d_in[25];
  const float* mlp_ln_g = (const float*)d_in[26];
  const float* mlp_ln_b = (const float*)d_in[27];
  const float* mlp_W2 = (const float*)d_in[28];
  const float* mlp_b2 = (const float*)d_in[29];
  const float* log_attn_w = (const float*)d_in[30];
  const float* log_mlp_w  = (const float*)d_in[31];
  const int*   topk = (const int*)d_in[32];

  const int NS = cB * cLS;           // 2048 spatial rows
  const int NG = cB * cG;            // 64 global rows

  char* ws = (char*)d_ws; size_t off = 0;
  auto alloc = [&](size_t bytes) -> void* {
    void* r = ws + off; off = (off + bytes + 255) & ~(size_t)255; return r;
  };
  // f16 activations & packed weights (pack uses first cD rows of Wk_l/Wv_l)
  _Float16* s_n_h  = (_Float16*)alloc((size_t)NS * cD * 2);
  _Float16* g_n_h  = (_Float16*)alloc((size_t)NG * cD * 2);
  _Float16* Wq_l_p = (_Float16*)alloc((size_t)cD * cINNER * 2);
  _Float16* Wk_l_p = (_Float16*)alloc((size_t)cD * cINNER * 2);
  _Float16* Wv_l_p = (_Float16*)alloc((size_t)cD * cINNER * 2);
  _Float16* Wo_l_p = (_Float16*)alloc((size_t)cINNER * cD * 2);
  _Float16* Wq_g_p = (_Float16*)alloc((size_t)cD * cINNER * 2);
  _Float16* Wk_g_p = (_Float16*)alloc((size_t)cD * cINNER * 2);
  _Float16* Wv_g_p = (_Float16*)alloc((size_t)cD * cINNER * 2);
  _Float16* Wo_g_p = (_Float16*)alloc((size_t)cINNER * cD * 2);
  _Float16* W1_p   = (_Float16*)alloc((size_t)cD * cHID * 2);
  // f32 intermediates
  float* Qf   = (float*)alloc((size_t)NS * cINNER * 4);
  float* Tq   = (float*)alloc((size_t)NS * cH * cPE * 4);
  float* Ksf  = (float*)alloc((size_t)NS * cINNER * 4);
  float* Vsf  = (float*)alloc((size_t)NS * cINNER * 4);
  float* Kglf = (float*)alloc((size_t)NG * cINNER * 4);
  float* Vglf = (float*)alloc((size_t)NG * cINNER * 4);
  float* Ksgf = (float*)alloc((size_t)NS * cINNER * 4);
  float* Vsgf = (float*)alloc((size_t)NS * cINNER * 4);
  float* Kggf = (float*)alloc((size_t)NG * cINNER * 4);
  float* Vggf = (float*)alloc((size_t)NG * cINNER * 4);
  float* Qgf  = (float*)alloc((size_t)NG * cINNER * 4);
  _Float16* outh = (_Float16*)alloc((size_t)NS * cINNER * 2);
  _Float16* ogh  = (_Float16*)alloc((size_t)NG * cINNER * 2);
  _Float16* s2h  = (_Float16*)alloc((size_t)NS * cD * 2);
  float* h1pre   = (float*)alloc((size_t)NS * cHID * 4);
  float* adisp   = (float*)alloc((size_t)NS * 2 * 4);

  float* out_spatial2 = (float*)d_out;                        // B*LS*D
  float* out_global2  = out_spatial2 + (size_t)NS * cD;       // B*G*D
  float* out_disp     = out_global2 + (size_t)NG * cD;        // B*LS*2

  // ---- weight conversion + fragment repack
  auto pack = [&](const float* W, _Float16* P, int Kd, int N) {
    int n = Kd * N;
    pack_w_kernel<<<(n + 255) / 256, 256, 0, stream>>>(W, P, Kd, N);
  };
  pack(Wq_l, Wq_l_p, cD, cINNER);
  pack(Wk_l, Wk_l_p, cD, cINNER);     // feature rows only (0..cD-1)
  pack(Wv_l, Wv_l_p, cD, cINNER);
  pack(Wo_l, Wo_l_p, cINNER, cD);
  pack(Wq_g, Wq_g_p, cD, cINNER);
  pack(Wk_g, Wk_g_p, cD, cINNER);
  pack(Wv_g, Wv_g_p, cD, cINNER);
  pack(Wo_g, Wo_g_p, cINNER, cD);
  pack(mlp_W1, W1_p, cD, cHID);

  // ---- layernorms -> f16 activations
  ln_f16_kernel<<<NS, 256, 0, stream>>>(spatial, ln_s_g, ln_s_b, s_n_h, cD);
  ln_f16_kernel<<<NG, 256, 0, stream>>>(global_latents, ln_g_g, ln_g_b, g_n_h, cD);

  // ---- WMMA projections
  launch_gemm(stream, s_n_h, cD, Wq_l_p, cINNER, Qf,   cINNER, NS, cINNER, cD, nullptr, nullptr, 0, nullptr, 0);
  launch_gemm(stream, s_n_h, cD, Wk_l_p, cINNER, Ksf,  cINNER, NS, cINNER, cD, nullptr, nullptr, 0, nullptr, 0);
  launch_gemm(stream, s_n_h, cD, Wv_l_p, cINNER, Vsf,  cINNER, NS, cINNER, cD, nullptr, nullptr, 0, nullptr, 0);
  launch_gemm(stream, g_n_h, cD, Wk_l_p, cINNER, Kglf, cINNER, NG, cINNER, cD, nullptr, nullptr, 0, nullptr, 0);
  launch_gemm(stream, g_n_h, cD, Wv_l_p, cINNER, Vglf, cINNER, NG, cINNER, cD, nullptr, nullptr, 0, nullptr, 0);
  launch_gemm(stream, g_n_h, cD, Wq_g_p, cINNER, Qgf,  cINNER, NG, cINNER, cD, nullptr, nullptr, 0, nullptr, 0);
  launch_gemm(stream, s_n_h, cD, Wk_g_p, cINNER, Ksgf, cINNER, NS, cINNER, cD, nullptr, nullptr, 0, nullptr, 0);
  launch_gemm(stream, s_n_h, cD, Wv_g_p, cINNER, Vsgf, cINNER, NS, cINNER, cD, nullptr, nullptr, 0, nullptr, 0);
  launch_gemm(stream, g_n_h, cD, Wk_g_p, cINNER, Kggf, cINNER, NG, cINNER, cD, nullptr, nullptr, 0, nullptr, 0);
  launch_gemm(stream, g_n_h, cD, Wv_g_p, cINNER, Vggf, cINNER, NG, cINNER, cD, nullptr, nullptr, 0, nullptr, 0);

  // ---- fold RPE K-projection onto Q side
  const float* Wk_rpe = Wk_l + (size_t)cD * cINNER;   // rows D..D+PE-1
  const float* Wv_rpe = Wv_l + (size_t)cD * cINNER;
  tq_kernel<<<NS, 512, 0, stream>>>(Qf, Wk_rpe, Tq);

  // ---- fused local attention (+ neighbor displacement)
  local_attn_kernel<<<NS, 256, 0, stream>>>(
      Qf, Tq, Ksf, Vsf, Kglf, Vglf, self_rpe, rpe, distances, topk,
      log_sigma, gb_l, head_weights, positions, Wv_rpe, outh, adisp);

  // ---- global attention (online softmax, 1 wave per (b,h))
  global_attn_kernel<<<cB * cH, 32, 0, stream>>>(Qgf, Ksgf, Kggf, Vsgf, Vggf,
                                                 gb_g, ogh);

  // ---- output projections with bias + residual
  launch_gemm(stream, outh, cINNER, Wo_l_p, cD, out_spatial2, cD,
              NS, cD, cINNER, bo_l, spatial, cD, s2h, cD);
  launch_gemm(stream, ogh, cINNER, Wo_g_p, cD, out_global2, cD,
              NG, cD, cINNER, bo_g, global_latents, cD, nullptr, 0);

  // ---- MLP first GEMM then fused gelu/LN/W2/combine/tanh
  launch_gemm(stream, s2h, cD, W1_p, cHID, h1pre, cHID,
              NS, cHID, cD, mlp_b1, nullptr, 0, nullptr, 0);
  mlp_disp_kernel<<<NS, cHID, 0, stream>>>(h1pre, mlp_ln_g, mlp_ln_b, mlp_W2,
                                           mlp_b2, adisp, log_attn_w,
                                           log_mlp_w, out_disp);
}
